// SingleLayerLstm_20813411517096
// MI455X (gfx1250) — matrix operand order repaired and checked by reading, not compiled, not run on hardware
//
#include <hip/hip_runtime.h>
#include <hip/hip_bf16.h>
#include <string.h>

// ---------------- types ----------------
typedef __attribute__((ext_vector_type(16))) __bf16 v16bf;
typedef __attribute__((ext_vector_type(8)))  float  v8f;

#define T_STEPS 256
#define BATCH   64
#define DHID    1024
#define DIN     1024
#define NGATE   4096   // 4*DHID
#define MTOT    16384  // T_STEPS*BATCH

__device__ __forceinline__ unsigned short f32_to_bf16_rne(float f) {
    unsigned int u = __float_as_uint(f);
    u += 0x7FFFu + ((u >> 16) & 1u);
    return (unsigned short)(u >> 16);
}

__device__ __forceinline__ float sigm(float x) {
    return 1.0f / (1.0f + __expf(-x));
}

// A fragment: 16x32 bf16, rows row0..row0+15 from LDS (row-major, stride lda u16).
// ISA 16-bit A layout: lanes 0-15 row M=lane, K=0..7 (v0-3) & 16..23 (v4-7);
// lanes 16-31 same rows, K=+8.  -> two contiguous 16B LDS loads per lane.
__device__ __forceinline__ v16bf load_frag_a(const unsigned short* lds, int row0, int lda) {
    int lane = threadIdx.x & 31;
    int r  = row0 + (lane & 15);
    int kh = (lane >> 4) * 8;
    union { v16bf v; uint4 u[2]; } f;
    f.u[0] = *(const uint4*)(lds + r * lda + kh);
    f.u[1] = *(const uint4*)(lds + r * lda + kh + 16);
    return f.v;
}

// B fragment: 32x16 bf16 from LDS stored TRANSPOSED [col][k] (stride ldb u16).
// ISA 16-bit B layout: lanes 0-15 col N=lane, K=0..7 & 16..23; lanes 16-31 K=+8.
__device__ __forceinline__ v16bf load_frag_b(const unsigned short* ldsT, int col0, int ldb) {
    int lane = threadIdx.x & 31;
    int c  = col0 + (lane & 15);
    int kb = (lane >> 4) * 8;
    union { v16bf v; uint4 u[2]; } f;
    f.u[0] = *(const uint4*)(ldsT + c * ldb + kb);
    f.u[1] = *(const uint4*)(ldsT + c * ldb + kb + 16);
    return f.v;
}

// ---------------- fp32 -> bf16 convert ----------------
__global__ void cvt_bf16_kernel(const float* __restrict__ src,
                                unsigned short* __restrict__ dst, int n) {
    int i = blockIdx.x * blockDim.x + threadIdx.x;
    int stride = gridDim.x * blockDim.x;
    for (; i < n; i += stride) dst[i] = f32_to_bf16_rne(src[i]);
}

// ---------------- fp32 [K][N] -> bf16 transposed [N][K] (LDS-tiled) ----------------
__global__ void __launch_bounds__(256)
cvt_transpose_kernel(const float* __restrict__ src, unsigned short* __restrict__ dst,
                     int K, int N) {
    __shared__ unsigned short tile[32][33];
    const int kb = blockIdx.y * 32, nb = blockIdx.x * 32;
    const int tx = threadIdx.x & 31, ty = threadIdx.x >> 5;   // 32 x 8
    #pragma unroll
    for (int i = 0; i < 32; i += 8)
        tile[ty + i][tx] = f32_to_bf16_rne(src[(size_t)(kb + ty + i) * N + nb + tx]);
    __syncthreads();
    #pragma unroll
    for (int i = 0; i < 32; i += 8)
        dst[(size_t)(nb + ty + i) * K + kb + tx] = tile[tx][ty + i];
}

// ---------------- intern = X @ Wf + bf  (M=16384, K=1024, N=4096) ----------------
// block: 256 thr (8 waves). Output tile 128(M) x 64(N). Wave w: rows w*16..+15, all 64 cols.
// WfT is pre-transposed bf16 [N][K] so B staging is contiguous b128 both sides.
__global__ void __launch_bounds__(256)
gemm_intern_kernel(const unsigned short* __restrict__ Xb,
                   const unsigned short* __restrict__ WfT,
                   const float* __restrict__ bias,
                   float* __restrict__ intern) {
    __shared__ unsigned short ldsA[128 * 32];   // [row][k]
    __shared__ unsigned short ldsBt[64 * 32];   // [col][k]

    const int tid = threadIdx.x;
    const int w   = tid >> 5;
    const int n0  = blockIdx.x * 64;
    const int m0  = blockIdx.y * 128;

    v8f acc0 = 0.0f, acc1 = 0.0f, acc2 = 0.0f, acc3 = 0.0f;

    for (int kc = 0; kc < DIN; kc += 32) {
        __syncthreads();
        // stage A: 128 rows x 32 k  (512 uint4 slots, 2 per thread)
        #pragma unroll
        for (int s = tid; s < 512; s += 256) {
            int r = s >> 2, seg = s & 3;
            *(uint4*)(ldsA + r * 32 + seg * 8) =
                *(const uint4*)(Xb + (size_t)(m0 + r) * DIN + kc + seg * 8);
        }
        // stage B^T: 64 n-rows x 32 k, contiguous both sides
        {
            int r = tid >> 2, seg = tid & 3;
            *(uint4*)(ldsBt + r * 32 + seg * 8) =
                *(const uint4*)(WfT + (size_t)(n0 + r) * DIN + kc + seg * 8);
        }
        __syncthreads();

        v16bf af = load_frag_a(ldsA, w * 16, 32);
        v16bf b0 = load_frag_b(ldsBt,  0, 32);
        v16bf b1 = load_frag_b(ldsBt, 16, 32);
        v16bf b2 = load_frag_b(ldsBt, 32, 32);
        v16bf b3 = load_frag_b(ldsBt, 48, 32);
        acc0 = __builtin_amdgcn_wmma_f32_16x16x32_bf16(false, af, false, b0, (short)0, acc0, false, false);
        acc1 = __builtin_amdgcn_wmma_f32_16x16x32_bf16(false, af, false, b1, (short)0, acc1, false, false);
        acc2 = __builtin_amdgcn_wmma_f32_16x16x32_bf16(false, af, false, b2, (short)0, acc2, false, false);
        acc3 = __builtin_amdgcn_wmma_f32_16x16x32_bf16(false, af, false, b3, (short)0, acc3, false, false);
    }

    // store: C tile layout — vgpr v: lanes0-15 M=v, lanes16-31 M=v+8; N=lane&15
    const int lane  = tid & 31;
    const int rbase = m0 + w * 16 + ((lane >> 4) * 8);
    v8f accs[4] = {acc0, acc1, acc2, acc3};
    #pragma unroll
    for (int nt = 0; nt < 4; ++nt) {
        int col = n0 + nt * 16 + (lane & 15);
        float b = bias[col];
        #pragma unroll
        for (int v = 0; v < 8; ++v)
            intern[(size_t)(rbase + v) * NGATE + col] = accs[nt][v] + b;
    }
}

// ---------------- persistent LSTM scan ----------------
// 64 WGs x 256 thr. WG g owns hidden cols [g*16, g*16+16). 16 z-tiles/WG (4 gates x 4 row-tiles).
// WrT is pre-transposed bf16 [NGATE][DHID].
__global__ void __launch_bounds__(256)
lstm_scan_kernel(const unsigned short* __restrict__ WrT,
                 const float* __restrict__ intern,
                 const float* __restrict__ c0,
                 const float* __restrict__ imask,
                 unsigned short* __restrict__ ybuf,   // 2 buffers of 64*1024 u16
                 float* __restrict__ Y,
                 float* __restrict__ C,
                 float* __restrict__ dlast,
                 unsigned int* __restrict__ bar) {
    __shared__ unsigned short ldsA[64 * 32];        // y chunk [row][k]
    __shared__ unsigned short ldsB[4 * 16 * 32];    // [gate*16+col][k]
    __shared__ float zbuf[4 * 64 * 16];             // [gate][m][n]

    const int tid  = threadIdx.x;
    const int w    = tid >> 5;
    const int lane = tid & 31;
    const int n0   = blockIdx.x * 16;

    const int t0 = w * 2, t1 = w * 2 + 1;
    const int q0 = t0 >> 2, r0 = t0 & 3;
    const int q1 = t1 >> 2, r1 = t1 & 3;

    // persistent cell state: 4 elements / thread, e = i*256 + tid, m=e>>4, n=e&15
    float creg[4];
    #pragma unroll
    for (int i = 0; i < 4; ++i) {
        int e = i * 256 + tid;
        creg[i] = c0[(e >> 4) * DHID + n0 + (e & 15)];
    }

    for (int t = 0; t < T_STEPS; ++t) {
        const unsigned short* cur = ybuf + (size_t)(t & 1) * (BATCH * DHID);
        unsigned short*       nxt = ybuf + (size_t)((t + 1) & 1) * (BATCH * DHID);

        v8f acc0 = 0.0f, acc1 = 0.0f;

        for (int kc = 0; kc < DHID; kc += 32) {
            __syncthreads();
            // stage A via CDNA5 async copy: y_prev 64 rows x 32 k (256 x b128)
            {
                int r = tid >> 2, seg = tid & 3;
                unsigned lo = (unsigned)(uintptr_t)(ldsA + r * 32 + seg * 8);
                const unsigned short* g = cur + (size_t)r * DHID + kc + seg * 8;
                asm volatile("global_load_async_to_lds_b128 %0, %1, off"
                             :: "v"(lo), "v"(g) : "memory");
            }
            // stage B^T: 4 gates x 16 cols x 32 k, contiguous both sides
            {
                int rr = tid >> 2, seg = tid & 3;     // rr: 0..63 -> (gate, col)
                int q = rr >> 4, c = rr & 15;
                const unsigned short* gsrc =
                    WrT + (size_t)(q * DHID + n0 + c) * DHID + kc + seg * 8;
                *(uint4*)(ldsB + (q * 16 + c) * 32 + seg * 8) = *(const uint4*)gsrc;
                if (kc + 32 < DHID)
                    __builtin_prefetch(gsrc + 32, 0, 3);   // global_prefetch_b8
            }
            asm volatile("s_wait_asynccnt 0x0" ::: "memory");
            __syncthreads();

            v16bf a0 = load_frag_a(ldsA, r0 * 16, 32);
            v16bf b0 = load_frag_b(ldsB + q0 * 16 * 32, 0, 32);
            acc0 = __builtin_amdgcn_wmma_f32_16x16x32_bf16(false, a0, false, b0, (short)0, acc0, false, false);
            v16bf a1 = load_frag_a(ldsA, r1 * 16, 32);
            v16bf b1 = load_frag_b(ldsB + q1 * 16 * 32, 0, 32);
            acc1 = __builtin_amdgcn_wmma_f32_16x16x32_bf16(false, a1, false, b1, (short)0, acc1, false, false);
        }

        // z tiles -> LDS
        {
            int cn = lane & 15, mo = (lane >> 4) * 8;
            #pragma unroll
            for (int v = 0; v < 8; ++v) {
                zbuf[(q0 * 64 + r0 * 16 + mo + v) * 16 + cn] = acc0[v];
                zbuf[(q1 * 64 + r1 * 16 + mo + v) * 16 + cn] = acc1[v];
            }
        }
        __syncthreads();

        // gates + state update
        const float* internT = intern + (size_t)t * BATCH * NGATE;
        #pragma unroll
        for (int i = 0; i < 4; ++i) {
            int e = i * 256 + tid;
            int m = e >> 4, n = e & 15;
            float z0 = zbuf[(0 * 64 + m) * 16 + n] + internT[(size_t)m * NGATE + 0 * DHID + n0 + n];
            float z1 = zbuf[(1 * 64 + m) * 16 + n] + internT[(size_t)m * NGATE + 1 * DHID + n0 + n];
            float z2 = zbuf[(2 * 64 + m) * 16 + n] + internT[(size_t)m * NGATE + 2 * DHID + n0 + n];
            float z3 = zbuf[(3 * 64 + m) * 16 + n] + internT[(size_t)m * NGATE + 3 * DHID + n0 + n];
            float mt = imask[t * BATCH + m];
            float gi = sigm(z0), gf = sigm(z1), go = sigm(z2);
            float ci = tanhf(z3);
            float cold = creg[i];
            float cnew = (gi * ci + gf * cold) * mt + cold * (1.0f - mt);
            float ynew = go * tanhf(cnew) * mt;
            creg[i] = cnew;
            size_t off = (size_t)t * (BATCH * DHID) + (size_t)m * DHID + n0 + n;
            Y[off] = ynew;
            C[off] = cnew;
            nxt[(size_t)m * DHID + n0 + n] = f32_to_bf16_rne(ynew);
        }

        // ---- grid barrier (monotonic counter; all 64 WGs co-resident) ----
        __syncthreads();
        if (tid == 0) {
            __threadfence();
            atomicAdd(bar, 1u);
            unsigned int target = (unsigned int)gridDim.x * (unsigned int)(t + 1);
            while (__hip_atomic_load(bar, __ATOMIC_ACQUIRE, __HIP_MEMORY_SCOPE_AGENT) < target)
                __builtin_amdgcn_s_sleep(2);
            __threadfence();
        }
        __syncthreads();
    }

    // final cell state
    #pragma unroll
    for (int i = 0; i < 4; ++i) {
        int e = i * 256 + tid;
        dlast[(size_t)(e >> 4) * DHID + n0 + (e & 15)] = creg[i];
    }
}

// ---------------- host launcher ----------------
extern "C" void kernel_launch(void* const* d_in, const int* in_sizes, int n_in,
                              void* d_out, int out_size, void* d_ws, size_t ws_size,
                              hipStream_t stream) {
    const float* X   = (const float*)d_in[0];   // (256,64,1024)
    const float* h0  = (const float*)d_in[1];   // (64,1024)
    const float* c0  = (const float*)d_in[2];   // (64,1024)
    const float* im  = (const float*)d_in[3];   // (256,64)
    const float* Wf  = (const float*)d_in[4];   // (1024,4096)
    const float* Wr  = (const float*)d_in[5];   // (1024,4096)
    const float* bf  = (const float*)d_in[6];   // (4096,)

    char* ws = (char*)d_ws;
    unsigned short* Xb   = (unsigned short*)(ws);                       // 32 MB
    unsigned short* WfT  = (unsigned short*)(ws + 33554432);            // 8 MB  [4096][1024]
    unsigned short* WrT  = (unsigned short*)(ws + 41943040);            // 8 MB  [4096][1024]
    unsigned short* yb   = (unsigned short*)(ws + 50331648);            // 256 KB (2 ping-pong)
    unsigned int*   bar  = (unsigned int*)  (ws + 50593792);            // 256 B
    float*          intern = (float*)       (ws + 50594048);            // 256 MB

    float* Yout = (float*)d_out;
    float* Cout = Yout + (size_t)T_STEPS * BATCH * DHID;
    float* dout = Cout + (size_t)T_STEPS * BATCH * DHID;

    hipMemsetAsync(bar, 0, 256, stream);

    cvt_bf16_kernel<<<1024, 256, 0, stream>>>(X,  Xb,  MTOT * DIN);
    cvt_bf16_kernel<<<64,   256, 0, stream>>>(h0, yb,  BATCH * DHID);  // ybuf[0] = h0

    dim3 tgrid(NGATE / 32, DIN / 32);   // (128, 32)
    cvt_transpose_kernel<<<tgrid, 256, 0, stream>>>(Wf, WfT, DIN,  NGATE);
    cvt_transpose_kernel<<<tgrid, 256, 0, stream>>>(Wr, WrT, DHID, NGATE);

    gemm_intern_kernel<<<dim3(NGATE / 64, MTOT / 128), 256, 0, stream>>>(Xb, WfT, bf, intern);

    lstm_scan_kernel<<<64, 256, 0, stream>>>(WrT, intern, c0, im, yb, Yout, Cout, dout, bar);
}
